// GCNForGraphEncoder_72576357367869
// MI455X (gfx1250) — matrix operand order repaired and checked by reading, not compiled, not run on hardware
//
#include <hip/hip_runtime.h>
#include <hip/hip_bf16.h>

// ---------------------------------------------------------------------------
// Types for CDNA5 WMMA
// ---------------------------------------------------------------------------
typedef __attribute__((ext_vector_type(16))) __bf16 v16bf;
typedef __attribute__((ext_vector_type(8)))  __bf16 v8bf;
typedef __attribute__((ext_vector_type(8)))  float  v8f;

#define HDIM 128   // D == H == 128
#define GQ   64    // graphs

// ---------------------------------------------------------------------------
// bf16 helpers: native casts (RNE); hi/lo split for f32-accurate WMMA
// ---------------------------------------------------------------------------
__device__ __forceinline__ void split_bf16(float f, __bf16& hi, __bf16& lo) {
    __bf16 h = (__bf16)f;
    hi = h;
    lo = (__bf16)(f - (float)h);
}

// ---------------------------------------------------------------------------
// Pack W[k][n] (row-major 128x128 f32) into per-wave B fragments, split hi/lo.
// Fragment (kb, nt): 32 lanes x 16 bf16.  Lane l (<16): column n = nt*16+l,
// K = kb*32 + 0..15 ; lane l+16: same column, K = kb*32 + 16..31.
// t = layer*16384 + (kb*8+nt)*512 + lane*16 + e
// ---------------------------------------------------------------------------
__global__ void wfrag_kernel(const float* __restrict__ W0,
                             const float* __restrict__ W1,
                             const float* __restrict__ W2,
                             const float* __restrict__ W3,
                             __bf16* __restrict__ hiF,
                             __bf16* __restrict__ loF) {
    int t = blockIdx.x * blockDim.x + threadIdx.x;
    if (t >= 4 * 16384) return;
    int e     = t & 15;
    int lane  = (t >> 4) & 31;
    int frag  = (t >> 9) & 31;
    int layer = t >> 14;
    int kb = frag >> 3, nt = frag & 7;
    int n = nt * 16 + (lane & 15);
    int k = kb * 32 + ((lane >> 4) * 16) + e;
    const float* W = (layer == 0) ? W0 : (layer == 1) ? W1 : (layer == 2) ? W2 : W3;
    __bf16 hb, lb;
    split_bf16(W[k * HDIM + n], hb, lb);
    hiF[t] = hb;
    loF[t] = lb;
}

// ---------------------------------------------------------------------------
// GEMM: Y[N,128] = relu?(X)[N,128] @ W[128,128] via split-bf16 WMMA.
// 256 threads = 8 waves. Block computes 16 rows x 128 cols.
// Phase 1: all threads convert the 16x128 f32 A tile -> bf16 hi/lo in LDS.
// Phase 2: wave nt computes its 16x16 tile; A frags come from LDS (B128 DS
//          loads), B frags from prepacked global fragments.
// ---------------------------------------------------------------------------
template <bool RELU_IN>
__global__ void __launch_bounds__(256)
gemm_bf16split(const float* __restrict__ X,
               const __bf16* __restrict__ WhiF,
               const __bf16* __restrict__ WloF,
               float* __restrict__ Y, int Nn) {
    __shared__ __bf16 sAhi[16 * HDIM];
    __shared__ __bf16 sAlo[16 * HDIM];

    const int rowB = blockIdx.x * 16;

    // ---- Phase 1: cooperative convert of A tile (8 f32 per thread) ----
    {
        const int base = threadIdx.x * 8;          // 0..2040
        int row = rowB + (base >> 7);
        if (row > Nn - 1) row = Nn - 1;            // clamp (N % 16 == 0 normally)
        const int kc = base & (HDIM - 1);
        const float4* src = (const float4*)(X + (size_t)row * HDIM + kc);
        float4 f0 = src[0], f1 = src[1];
        float a[8] = {f0.x, f0.y, f0.z, f0.w, f1.x, f1.y, f1.z, f1.w};
        v8bf hi, lo;
#pragma unroll
        for (int i = 0; i < 8; ++i) {
            float v = RELU_IN ? fmaxf(a[i], 0.0f) : a[i];
            __bf16 hb, lb;
            split_bf16(v, hb, lb);
            hi[i] = hb;
            lo[i] = lb;
        }
        *(v8bf*)(sAhi + base) = hi;
        *(v8bf*)(sAlo + base) = lo;
    }
    __syncthreads();

    // ---- Phase 2: WMMA ----
    const int lane  = threadIdx.x & 31;
    const int nt    = threadIdx.x >> 5;            // column tile 0..7
    const int m     = lane & 15;
    const int khalf = lane >> 4;

    const __bf16* ph = sAhi + m * HDIM + khalf * 16;
    const __bf16* pl = sAlo + m * HDIM + khalf * 16;

    v8f acc = {0.f, 0.f, 0.f, 0.f, 0.f, 0.f, 0.f, 0.f};

#pragma unroll
    for (int kb = 0; kb < 4; ++kb) {
        // A fragment: elements 0..15 -> K = kb*32 + khalf*16 + i
        v8bf h0 = *(const v8bf*)(ph + kb * 32);
        v8bf h1 = *(const v8bf*)(ph + kb * 32 + 8);
        v8bf l0 = *(const v8bf*)(pl + kb * 32);
        v8bf l1 = *(const v8bf*)(pl + kb * 32 + 8);
        v16bf ahi = __builtin_shufflevector(h0, h1, 0, 1, 2, 3, 4, 5, 6, 7,
                                            8, 9, 10, 11, 12, 13, 14, 15);
        v16bf alo = __builtin_shufflevector(l0, l1, 0, 1, 2, 3, 4, 5, 6, 7,
                                            8, 9, 10, 11, 12, 13, 14, 15);
        const int frag = kb * 8 + nt;
        const v16bf bhi = *(const v16bf*)(WhiF + frag * 512 + lane * 16);
        const v16bf blo = *(const v16bf*)(WloF + frag * 512 + lane * 16);
        acc = __builtin_amdgcn_wmma_f32_16x16x32_bf16(false, ahi, false, bhi,
                                                      (short)0, acc, false, false);
        acc = __builtin_amdgcn_wmma_f32_16x16x32_bf16(false, alo, false, bhi,
                                                      (short)0, acc, false, false);
        acc = __builtin_amdgcn_wmma_f32_16x16x32_bf16(false, ahi, false, blo,
                                                      (short)0, acc, false, false);
    }
    // C/D layout: VGPR r, lanes 0-15 -> M=r ; lanes 16-31 -> M=r+8 ; N = lane&15
    const int n    = nt * 16 + (lane & 15);
    const int mOff = (lane < 16) ? 0 : 8;
#pragma unroll
    for (int r = 0; r < 8; ++r) {
        int row = rowB + r + mOff;
        if (row < Nn) Y[(size_t)row * HDIM + n] = acc[r];
    }
}

// ---------------------------------------------------------------------------
// degree / norm
// ---------------------------------------------------------------------------
__global__ void init_deg_kernel(float* __restrict__ deg, int Nn) {
    int i = blockIdx.x * blockDim.x + threadIdx.x;
    if (i < Nn) deg[i] = 1.0f;   // self-loop contributes 1
}
__global__ void deg_kernel(const int* __restrict__ cols, float* __restrict__ deg, int E) {
    int e = blockIdx.x * blockDim.x + threadIdx.x;
    if (e < E) atomicAdd(&deg[cols[e]], 1.0f);
}
__global__ void rsqrt_kernel(float* __restrict__ deg, int Nn) {
    int i = blockIdx.x * blockDim.x + threadIdx.x;
    if (i < Nn) {
        float d = deg[i];
        deg[i] = (d > 0.0f) ? rsqrtf(d) : 0.0f;
    }
}

// agg[i][j] = b[j]
__global__ void init_bias_kernel(float* __restrict__ agg, const float* __restrict__ b, int total) {
    int i = blockIdx.x * blockDim.x + threadIdx.x;
    if (i < total) agg[i] = b[i & (HDIM - 1)];
}

// ---------------------------------------------------------------------------
// Edge scatter: one wave per edge (incl. N self-loops), float4 per lane,
// 4x global_atomic_add_f32.
// ---------------------------------------------------------------------------
__global__ void __launch_bounds__(256)
scatter_kernel(const float* __restrict__ Hsrc,
               const int* __restrict__ rows,
               const int* __restrict__ cols,
               const float* __restrict__ dinv,
               float* __restrict__ agg, int E, int Nn) {
    int wave = (int)((blockIdx.x * (unsigned)blockDim.x + threadIdx.x) >> 5);
    int lane = threadIdx.x & 31;
    int total = E + Nn;
    if (wave >= total) return;
    int r, c;
    if (wave < E) { r = rows[wave]; c = cols[wave]; }
    else          { r = c = wave - E; }
    float nrm = dinv[r] * dinv[c];
    const float4 v = ((const float4*)(Hsrc + (size_t)r * HDIM))[lane];
    float* dst = agg + (size_t)c * HDIM + lane * 4;
    atomicAdd(dst + 0, v.x * nrm);
    atomicAdd(dst + 1, v.y * nrm);
    atomicAdd(dst + 2, v.z * nrm);
    atomicAdd(dst + 3, v.w * nrm);
}

// ---------------------------------------------------------------------------
// Pooling: batch is sorted; each block handles 64 nodes x 128 feats with
// thread-local accumulation, flushing atomics only on graph-id change.
// Applies final ReLU while reading.
// ---------------------------------------------------------------------------
__global__ void zero_kernel(float* __restrict__ p, int total) {
    int i = blockIdx.x * blockDim.x + threadIdx.x;
    if (i < total) p[i] = 0.0f;
}
__global__ void pool_kernel(const float* __restrict__ Hn,
                            const int* __restrict__ batch,
                            float* __restrict__ sums,
                            float* __restrict__ counts, int Nn) {
    int j = threadIdx.x;                 // 0..127
    int i0 = blockIdx.x * 64;
    int i1 = i0 + 64; if (i1 > Nn) i1 = Nn;
    float accv = 0.0f, cnt = 0.0f;
    int curg = -1;
    for (int i = i0; i < i1; ++i) {
        int g = batch[i];
        if (g != curg) {
            if (curg >= 0) {
                atomicAdd(&sums[curg * HDIM + j], accv);
                if (j == 0) atomicAdd(&counts[curg], cnt);
            }
            curg = g; accv = 0.0f; cnt = 0.0f;
        }
        accv += fmaxf(Hn[(size_t)i * HDIM + j], 0.0f);
        cnt  += 1.0f;
    }
    if (curg >= 0) {
        atomicAdd(&sums[curg * HDIM + j], accv);
        if (j == 0) atomicAdd(&counts[curg], cnt);
    }
}
__global__ void divide_kernel(const float* __restrict__ sums,
                              const float* __restrict__ counts,
                              float* __restrict__ out) {
    int g = blockIdx.x, j = threadIdx.x;
    out[g * HDIM + j] = sums[g * HDIM + j] / fmaxf(counts[g], 1.0f);
}

// ---------------------------------------------------------------------------
extern "C" void kernel_launch(void* const* d_in, const int* in_sizes, int n_in,
                              void* d_out, int out_size, void* d_ws, size_t ws_size,
                              hipStream_t stream) {
    const float* x     = (const float*)d_in[0];
    const int*   ei    = (const int*)d_in[1];     // [2, E]
    const int*   batch = (const int*)d_in[2];
    const float* W[4]  = {(const float*)d_in[3], (const float*)d_in[5],
                          (const float*)d_in[7], (const float*)d_in[9]};
    const float* b[4]  = {(const float*)d_in[4], (const float*)d_in[6],
                          (const float*)d_in[8], (const float*)d_in[10]};
    const int E  = in_sizes[1] / 2;
    const int Nn = in_sizes[2];
    const int* rows = ei;
    const int* cols = ei + E;

    // --- carve workspace ---
    char* ws = (char*)d_ws;
    size_t off = 0;
    auto carve = [&](size_t bytes) -> void* {
        void* p = ws + off;
        off += (bytes + 255) & ~(size_t)255;
        return p;
    };
    float*  gemm_out = (float*)carve((size_t)Nn * HDIM * sizeof(float));
    float*  agg      = (float*)carve((size_t)Nn * HDIM * sizeof(float));
    float*  dinv     = (float*)carve((size_t)Nn * sizeof(float));
    __bf16* whi      = (__bf16*)carve(4 * 16384 * sizeof(__bf16));
    __bf16* wlo      = (__bf16*)carve(4 * 16384 * sizeof(__bf16));
    float*  sums     = (float*)carve((GQ * HDIM + GQ) * sizeof(float));
    float*  counts   = sums + GQ * HDIM;
    (void)ws_size; (void)n_in;

    // --- weight fragments (all 4 layers) ---
    wfrag_kernel<<<(4 * 16384 + 255) / 256, 256, 0, stream>>>(W[0], W[1], W[2], W[3], whi, wlo);

    // --- degree -> dinv ---
    init_deg_kernel<<<(Nn + 255) / 256, 256, 0, stream>>>(dinv, Nn);
    deg_kernel<<<(E + 255) / 256, 256, 0, stream>>>(cols, dinv, E);
    rsqrt_kernel<<<(Nn + 255) / 256, 256, 0, stream>>>(dinv, Nn);

    const int gemmGrid    = (Nn + 15) / 16;
    const int biasTotal   = Nn * HDIM;
    const long long sthreads = (long long)(E + Nn) * 32;
    const int scatGrid    = (int)((sthreads + 255) / 256);

    // --- 4 GCN layers ---
    for (int layer = 0; layer < 4; ++layer) {
        const __bf16* lhi = whi + layer * 16384;
        const __bf16* llo = wlo + layer * 16384;
        if (layer == 0)
            gemm_bf16split<false><<<gemmGrid, 256, 0, stream>>>(x, lhi, llo, gemm_out, Nn);
        else
            gemm_bf16split<true><<<gemmGrid, 256, 0, stream>>>(agg, lhi, llo, gemm_out, Nn);
        init_bias_kernel<<<(biasTotal + 255) / 256, 256, 0, stream>>>(agg, b[layer], biasTotal);
        scatter_kernel<<<scatGrid, 256, 0, stream>>>(gemm_out, rows, cols, dinv, agg, E, Nn);
        // agg now holds pre-ReLU layer output; ReLU fused into next consumer
    }

    // --- global mean pool (with final ReLU fused) ---
    zero_kernel<<<(GQ * HDIM + GQ + 255) / 256, 256, 0, stream>>>(sums, GQ * HDIM + GQ);
    pool_kernel<<<(Nn + 63) / 64, HDIM, 0, stream>>>(agg, batch, sums, counts, Nn);
    divide_kernel<<<GQ, HDIM, 0, stream>>>(sums, counts, (float*)d_out);
    (void)out_size;
}
